// GAT_73418170958308
// MI455X (gfx1250) — matrix operand order repaired and checked by reading, not compiled, not run on hardware
//
#include <hip/hip_runtime.h>
#include <hip/hip_bf16.h>

// GAT forward for MI455X (gfx1250), wave32 + bf16 WMMA.
// N=4096, NIN=128, NHID=64, H=8, NOUT=64.

#define GN    4096
#define GNIN  128
#define GH    8
#define GDH   64   // hidden per head == NOUT == 64

typedef __bf16 bf16;
typedef __attribute__((ext_vector_type(16))) __bf16 v16bf;
typedef __attribute__((ext_vector_type(8)))  float  v8f;

__device__ __forceinline__ __bf16 to_bf(float f)  { return (__bf16)f; }
__device__ __forceinline__ __bf16 to_bf(__bf16 b) { return b; }

__device__ __forceinline__ float leaky02(float x) { return fmaxf(x, 0.2f * x); }

// ---------------------------------------------------------------------------
// Projection: Wh = X[N,K] @ W[K,64]  (bf16 WMMA 16x16x32, f32 accum)
// Also emits Whb (bf16 row-major, the WMMA-B/V operand for attention) and
// src[n] = Wh[n,:]@a[:64], dst[n] = Wh[n,:]@a[64:128] via lane reductions.
// One wave owns one 16-row tile; 8 waves / 256-thread block.
// ---------------------------------------------------------------------------
template<typename XT>
__global__ __launch_bounds__(256)
void proj_kernel(const XT* __restrict__ X, int K, int ldx,
                 const float* __restrict__ Wbase, int wHeadStride,
                 const float* __restrict__ aBase, int aHeadStride,
                 bf16* __restrict__ WhbBase, int whbHeadStride,
                 float* __restrict__ srcBase, float* __restrict__ dstBase,
                 int sdHeadStride)
{
    const int head = blockIdx.y;
    const float* Wm   = Wbase   + (size_t)head * wHeadStride;
    const float* av   = aBase   + (size_t)head * aHeadStride;
    bf16*  Whb        = WhbBase + (size_t)head * whbHeadStride;
    float* srcv       = srcBase + (size_t)head * sdHeadStride;
    float* dstv       = dstBase + (size_t)head * sdHeadStride;

    const int lane = threadIdx.x & 31;
    const int wave = threadIdx.x >> 5;
    const int hi   = lane >> 4;        // K-half select (A/B layout)
    const int lq   = lane & 15;
    const int rowbase = (blockIdx.x * 8 + wave) * 16;
    const int arow = rowbase + lq;     // A-operand row for this lane

    v8f acc[4] = {};                   // 16 x 64 f32 accumulator (4 N-tiles)

    for (int kb = 0; kb < K; kb += 32) {
        // A tile 16x32 bf16: lane holds row `arow`, two runs of 8 contiguous K.
        v16bf A;
#pragma unroll
        for (int u = 0; u < 2; ++u) {
            const XT* xp = X + (size_t)arow * ldx + kb + u * 16 + hi * 8;
#pragma unroll
            for (int i = 0; i < 8; ++i) A[u * 8 + i] = to_bf(xp[i]);
        }
        // B tiles 32x16: lane holds col lq, K rows kb + hi*8 + {0..7, 16..23}.
        const int krow = kb + hi * 8;
        const float* wp = Wm + (size_t)krow * 64 + lq;
#pragma unroll
        for (int t = 0; t < 4; ++t) {
            v16bf B;
#pragma unroll
            for (int i = 0; i < 16; ++i) B[i] = to_bf(wp[((i >> 3) * 16 + (i & 7)) * 64 + t * 16]);
            acc[t] = __builtin_amdgcn_wmma_f32_16x16x32_bf16(
                false, A, false, B, (short)0, acc[t], false, false);
        }
    }

    // C layout: vgpr r -> row rowbase + r + hi*8, col = t*16 + lq.
#pragma unroll
    for (int r = 0; r < 8; ++r) {
        const int orow = rowbase + r + hi * 8;
        float s = 0.f, d = 0.f;
#pragma unroll
        for (int t = 0; t < 4; ++t) {
            const float v = acc[t][r];
            const int col = t * 16 + lq;
            s += v * av[col];
            d += v * av[64 + col];
            Whb[(size_t)orow * 64 + col] = (bf16)v;
        }
        // Reduce dot-partials over the 16 lanes of each half-wave.
#pragma unroll
        for (int m = 1; m < 16; m <<= 1) {
            s += __shfl_xor(s, m, 32);
            d += __shfl_xor(d, m, 32);
        }
        if (lq == 0) { srcv[orow] = s; dstv[orow] = d; }
    }
}

// ---------------------------------------------------------------------------
// dmax[h] = max_j dst[h][j]  (gives an exact-safe softmax shift since
// leaky_relu is monotone: all unmasked scores <= leaky(src_i + dmax)).
// ---------------------------------------------------------------------------
__global__ __launch_bounds__(256)
void dmax_kernel(const float* __restrict__ dstv, int headStride,
                 float* __restrict__ dmax)
{
    const int head = blockIdx.x;
    const float* d = dstv + (size_t)head * headStride;
    const int tid = threadIdx.x;
    float m = -3.0e38f;
    for (int i = tid; i < GN; i += 256) m = fmaxf(m, d[i]);
    __shared__ float red[256];
    red[tid] = m;
    __syncthreads();
    for (int s = 128; s > 0; s >>= 1) {
        if (tid < s) red[tid] = fmaxf(red[tid], red[tid + s]);
        __syncthreads();
    }
    if (tid == 0) dmax[head] = red[0];
}

// ---------------------------------------------------------------------------
// Fused masked softmax + aggregation:  out = softmax(mask(leaky(s_i+d_j))) @ V
// Scores are built directly in the bf16 A-operand layout, V tiles load as
// contiguous 32B rows (WMMA B layout from row-major memory). No rescaling
// needed: exponents shifted by the per-row bound m_i, masked entries are 0.
// ---------------------------------------------------------------------------
template<bool ELU, typename OT>
__global__ __launch_bounds__(256)
void attn_kernel(const bf16* __restrict__ VbBase, int vHeadStride,
                 const float* __restrict__ srcBase,
                 const float* __restrict__ dstBase, int sdHeadStride,
                 const float* __restrict__ dmax,
                 const int* __restrict__ adj,
                 OT* __restrict__ out, int outStride, int headColStride)
{
    const int head = blockIdx.y;
    const bf16*  Vb   = VbBase  + (size_t)head * vHeadStride;
    const float* srcv = srcBase + (size_t)head * sdHeadStride;
    const float* dstv = dstBase + (size_t)head * sdHeadStride;

    const int lane = threadIdx.x & 31;
    const int wave = threadIdx.x >> 5;
    const int hi   = lane >> 4;
    const int lq   = lane & 15;
    const int rowbase = (blockIdx.x * 8 + wave) * 16;
    const int arow = rowbase + lq;

    const float L2E  = 1.4426950408889634f;
    const float s_i  = srcv[arow];
    const float m_i  = leaky02(s_i + dmax[head]);
    const float negm = -m_i * L2E;
    const int* adjrow = adj + (size_t)arow * GN;

    v8f   acc[4] = {};
    float rowsum = 0.f;

    for (int jb = 0; jb < GN; jb += 32) {
        // Build P (16x32) in A layout: lane covers row `arow`,
        // K pairs are contiguous -> dst/adj loads vectorize to runs of 8.
        v16bf A;
#pragma unroll
        for (int u = 0; u < 2; ++u) {
            const int base = jb + u * 16 + hi * 8;
#pragma unroll
            for (int i = 0; i < 8; ++i) {
                const float t = s_i + dstv[base + i];
                const float e = leaky02(t);
                const float p = (adjrow[base + i] != 0)
                    ? __builtin_amdgcn_exp2f(fmaf(e, L2E, negm)) : 0.f;
                rowsum += p;
                A[u * 8 + i] = (bf16)p;
            }
        }
        // V tiles (B operand): lane = col lq, K rows jb + hi*8 + {0..7, 16..23}.
        const int krow = jb + hi * 8;
        const bf16* vrow = Vb + (size_t)krow * 64 + lq;
#pragma unroll
        for (int t = 0; t < 4; ++t) {
            v16bf B;
#pragma unroll
            for (int i = 0; i < 16; ++i) B[i] = vrow[((i >> 3) * 16 + (i & 7)) * 64 + t * 16];
            acc[t] = __builtin_amdgcn_wmma_f32_16x16x32_bf16(
                false, A, false, B, (short)0, acc[t], false, false);
        }
    }

    // Combine the two K-halves of each row's sum, then broadcast into C layout.
    rowsum += __shfl_xor(rowsum, 16, 32);

    const int colbase = head * headColStride;
#pragma unroll
    for (int r = 0; r < 8; ++r) {
        const float rs  = __shfl(rowsum, r + hi * 8, 32);
        const float inv = (rs != 0.f) ? 1.f / rs : 0.f;
        const int orow  = rowbase + r + hi * 8;
#pragma unroll
        for (int t = 0; t < 4; ++t) {
            float v = acc[t][r] * inv;
            if (ELU) v = (v > 0.f) ? v : (__builtin_amdgcn_exp2f(v * L2E) - 1.f);
            out[(size_t)orow * outStride + colbase + t * 16 + lq] = (OT)v;
        }
    }
}

// ---------------------------------------------------------------------------
// BatchNorm (batch stats over nodes) + tanh. One block per feature column.
// ---------------------------------------------------------------------------
__global__ __launch_bounds__(256)
void bn_tanh_kernel(const float* __restrict__ pre,
                    const float* __restrict__ gamma,
                    const float* __restrict__ beta,
                    float* __restrict__ outp)
{
    const int c   = blockIdx.x;   // 0..63
    const int tid = threadIdx.x;
    float sum = 0.f, sq = 0.f;
    for (int i = tid; i < GN; i += 256) {
        const float v = pre[(size_t)i * 64 + c];
        sum += v; sq += v * v;
    }
    __shared__ float s1[256], s2[256];
    s1[tid] = sum; s2[tid] = sq;
    __syncthreads();
    for (int s = 128; s > 0; s >>= 1) {
        if (tid < s) { s1[tid] += s1[tid + s]; s2[tid] += s2[tid + s]; }
        __syncthreads();
    }
    const float mu    = s1[0] * (1.f / GN);
    const float var   = s2[0] * (1.f / GN) - mu * mu;
    const float scale = gamma[c] * rsqrtf(var + 1e-5f);
    const float shift = beta[c] - mu * scale;
    for (int i = tid; i < GN; i += 256) {
        const float v = pre[(size_t)i * 64 + c];
        outp[(size_t)i * 64 + c] = tanhf(fmaf(v, scale, shift));
    }
}

// ---------------------------------------------------------------------------
extern "C" void kernel_launch(void* const* d_in, const int* in_sizes, int n_in,
                              void* d_out, int out_size, void* d_ws, size_t ws_size,
                              hipStream_t stream)
{
    const float* x       = (const float*)d_in[0];  // [4096,128]
    const int*   adj     = (const int*)  d_in[1];  // [4096,4096]
    const float* W_heads = (const float*)d_in[2];  // [8,128,64]
    const float* a_heads = (const float*)d_in[3];  // [8,128]
    const float* W_out   = (const float*)d_in[4];  // [512,64]
    const float* a_out   = (const float*)d_in[5];  // [128]
    const float* gamma   = (const float*)d_in[6];  // [64]
    const float* beta    = (const float*)d_in[7];  // [64]
    float* outp = (float*)d_out;                   // [4096,64]

    // Workspace carve-up (~10.4 MB total); everything is written before read.
    char* ws = (char*)d_ws;
    size_t off = 0;
    auto carve = [&](size_t bytes) -> void* {
        void* p = ws + off;
        off = (off + bytes + 255) & ~(size_t)255;
        return p;
    };
    bf16*  Whb   = (bf16*) carve((size_t)GH * GN * 64 * sizeof(bf16)); // 4 MB, per-head V
    bf16*  hcatb = (bf16*) carve((size_t)GN * 512 * sizeof(bf16));     // 4 MB, ELU'd concat
    bf16*  Wh2b  = (bf16*) carve((size_t)GN * 64 * sizeof(bf16));      // 0.5 MB
    float* src1  = (float*)carve((size_t)GH * GN * sizeof(float));
    float* dst1  = (float*)carve((size_t)GH * GN * sizeof(float));
    float* src2  = (float*)carve((size_t)GN * sizeof(float));
    float* dst2  = (float*)carve((size_t)GN * sizeof(float));
    float* dmax1 = (float*)carve(256);
    float* dmax2 = (float*)carve(256);
    float* pre   = (float*)carve((size_t)GN * 64 * sizeof(float));     // 1 MB
    (void)ws_size; (void)n_in; (void)in_sizes; (void)out_size;

    const dim3 blk(256);

    // ---- Layer 1: per-head projection + masked-softmax attention (ELU out) ----
    proj_kernel<float><<<dim3(32, GH), blk, 0, stream>>>(
        x, GNIN, GNIN,
        W_heads, GNIN * 64,
        a_heads, 2 * GDH,
        Whb, GN * 64,
        src1, dst1, GN);

    dmax_kernel<<<dim3(GH), blk, 0, stream>>>(dst1, GN, dmax1);

    attn_kernel<true, bf16><<<dim3(32, GH), blk, 0, stream>>>(
        Whb, GN * 64,
        src1, dst1, GN,
        dmax1, adj,
        hcatb, GH * GDH /*row stride 512*/, GDH /*col offset per head*/);

    // ---- Layer 2: output attention over the concatenated features ----
    proj_kernel<bf16><<<dim3(32, 1), blk, 0, stream>>>(
        hcatb, GH * GDH, GH * GDH,
        W_out, 0,
        a_out, 0,
        Wh2b, 0,
        src2, dst2, 0);

    dmax_kernel<<<dim3(1), blk, 0, stream>>>(dst2, GN, dmax2);

    attn_kernel<false, float><<<dim3(32, 1), blk, 0, stream>>>(
        Wh2b, 0,
        src2, dst2, 0,
        dmax2, adj,
        pre, 64, 0);

    // ---- BatchNorm (batch stats) + tanh ----
    bn_tanh_kernel<<<dim3(64), blk, 0, stream>>>(pre, gamma, beta, outp);
}
